// Encoder_54614804136064
// MI455X (gfx1250) — compile-verified, matrix operand order checked by reading, not run on hardware
//
#include <hip/hip_runtime.h>
#include <hip/hip_bf16.h>
#include <math.h>

typedef __attribute__((ext_vector_type(16))) _Float16 v16h;
typedef __attribute__((ext_vector_type(8)))  float    v8f;

#define VOCAB   100000
#define D_IN    128
#define D_HID   256
#define D_LAT   128
#define BATCH   16384
#define KCTX    20
#define ROWTILES 2
#define BLOCK   128   // 4 waves of 32

// ---------------------------------------------------------------------------
// WMMA fragment layouts (CDNA5 ISA 7.12.2, wave32):
//  A (16x32 f16): lane L holds row M=L%16; its v16h elements map to
//     K = kt*32 + (e>>3)*16 + (L>=16 ? 8 : 0) + (e&7)
//  B (32x16 f16): symmetric, lane L holds column N=L%16, same K split.
//  C/D (16x16 f32): lane L holds column N=L%16; VGPR v holds row M=v+(L>=16?8:0)
// Everything is pre-swizzled in LDS so a fragment is 32 lanes x 16 contiguous
// halves: frag[lane*16 + e]  (1 KB per fragment, 32B per lane -> ds_load_b128)
// ---------------------------------------------------------------------------

__device__ __forceinline__ v16h ldsFrag(const _Float16* base, int lane) {
  return *(const v16h*)(base + lane * 16);
}

// Scatter 16 consecutive K values (kbase multiple of 16) of row m into a
// 4-kt fragment buffer (A-matrix layout).
__device__ __forceinline__ void scatterRow16(_Float16* dst, int m, int kbase,
                                             const float* v) {
  const int kt = kbase >> 5;
  const int g  = (kbase >> 4) & 1;          // which 16-K half of the kt tile
  _Float16* fb = dst + kt * 512 + g * 8;    // e base = g*8
#pragma unroll
  for (int j = 0; j < 16; ++j) {
    const int lane = m + ((j >> 3) << 4);   // +16 lanes for K%16 >= 8
    fb[lane * 16 + (j & 7)] = (_Float16)v[j];
  }
}

__global__ __launch_bounds__(BLOCK)
void enc_kernel(const int* __restrict__ center_ids,
                const int* __restrict__ context_ids,
                const float* __restrict__ emb,
                const float* __restrict__ Wf,
                const float* __restrict__ bf,
                const float* __restrict__ Wu,
                const float* __restrict__ bu,
                const float* __restrict__ Wv,
                const float* __restrict__ bv,
                float* __restrict__ mu_out,
                float* __restrict__ sig_out) {
  // LDS: 128KB Wf frags + 8KB ctx double buffer + 4KB center + 8.25KB h
  __shared__ alignas(32) _Float16 sWf[2 * 4 * 16 * 512]; // [mat][kt][nt][frag]
  __shared__ alignas(32) _Float16 sA[2 * 4 * 512];       // [buf][kt][frag]
  __shared__ alignas(32) _Float16 sCen[4 * 512];         // [kt][frag]
  __shared__ alignas(32) _Float16 sH[16 * 264];          // h rows, stride 264

  const int tid  = threadIdx.x;
  const int lane = tid & 31;
  const int wq   = tid >> 5;                // wave id 0..3 -> N range wq*64
  const v8f zero = {0.f,0.f,0.f,0.f,0.f,0.f,0.f,0.f};

  // ---- stage Wf (f32 -> f16 B-fragments), once per block ----
  // Chunk = (mat, K, 16-wide N tile): 64B coalesced load, constant e/laneHi.
  for (int i = tid; i < 2 * D_IN * 16; i += BLOCK) {   // 4096 chunks
    const int mat = i >> 11;
    const int rem = i & 2047;
    const int Kv  = rem >> 4;               // 0..127
    const int nt  = rem & 15;               // N tile 0..15
    const float4* src =
        (const float4*)(Wf + (size_t)(mat * D_IN + Kv) * D_HID + nt * 16);
    float4 v[4] = { src[0], src[1], src[2], src[3] };
    const float* vv = (const float*)v;
    const int kt = Kv >> 5, K32 = Kv & 31, g = K32 >> 4, r = K32 & 15;
    const int laneHi = (r >> 3) << 4;
    const int e = g * 8 + (r & 7);
    _Float16* dst = sWf + ((((mat * 4 + kt) << 4) + nt) << 9) + e;
#pragma unroll
    for (int j = 0; j < 16; ++j) dst[(j + laneHi) * 16] = (_Float16)vv[j];
  }

  const int mrow = tid >> 3;        // staging: row 0..15
  const int c0   = (tid & 7) * 16;  // staging: K base (8 threads x 16 K = 128)

  float bfv[4];
#pragma unroll
  for (int q = 0; q < 4; ++q) bfv[q] = bf[wq * 64 + q * 16 + (lane & 15)];

  for (int rt = 0; rt < ROWTILES; ++rt) {
    const int b0 = (blockIdx.x * ROWTILES + rt) * 16;

    // ---- stage center embeddings (A fragments) ----
    {
      const int id = center_ids[b0 + mrow];
      const float4* s = (const float4*)(emb + (size_t)id * D_IN + c0);
      float4 v[4] = { s[0], s[1], s[2], s[3] };
      scatterRow16(sCen, mrow, c0, (const float*)v);
    }
    // ---- preload + stage context k = 0 ----
    float4 cv[4];
    {
      const int id = context_ids[(b0 + mrow) * KCTX + 0];
      const float4* s = (const float4*)(emb + (size_t)id * D_IN + c0);
      cv[0] = s[0]; cv[1] = s[1]; cv[2] = s[2]; cv[3] = s[3];
    }
    scatterRow16(sA, mrow, c0, (const float*)cv);
    __syncthreads();

    // ---- center projection: cen[q] = center_e @ Wf_c (this wave's 4 N tiles)
    v8f cen[4];
#pragma unroll
    for (int q = 0; q < 4; ++q) cen[q] = zero;
#pragma unroll
    for (int kt = 0; kt < 4; ++kt) {
      const v16h a = ldsFrag(sCen + kt * 512, lane);
#pragma unroll
      for (int q = 0; q < 4; ++q) {
        const v16h b = ldsFrag(sWf + ((((0 * 4 + kt) << 4) + (wq * 4 + q)) << 9), lane);
        cen[q] = __builtin_amdgcn_wmma_f32_16x16x32_f16(
            false, a, false, b, (short)0, cen[q], false, false);
      }
    }

    // ---- k loop: hsum += relu(ctx@Wf_x + cen + bf), double-buffered staging
    v8f hsum[4];
#pragma unroll
    for (int q = 0; q < 4; ++q) hsum[q] = zero;

    for (int k = 0; k < KCTX; ++k) {
      const _Float16* curA = sA + (k & 1) * 2048;
      const bool more = (k + 1 < KCTX);
      if (more) {  // issue next gather early to overlap with WMMA
        const int id = context_ids[(b0 + mrow) * KCTX + (k + 1)];
        const float4* s = (const float4*)(emb + (size_t)id * D_IN + c0);
        cv[0] = s[0]; cv[1] = s[1]; cv[2] = s[2]; cv[3] = s[3];
      }
      v8f acc[4];
#pragma unroll
      for (int q = 0; q < 4; ++q) acc[q] = zero;
#pragma unroll
      for (int kt = 0; kt < 4; ++kt) {
        const v16h a = ldsFrag(curA + kt * 512, lane);
#pragma unroll
        for (int q = 0; q < 4; ++q) {
          const v16h b = ldsFrag(sWf + ((((4 + kt) << 4) + (wq * 4 + q)) << 9), lane);
          acc[q] = __builtin_amdgcn_wmma_f32_16x16x32_f16(
              false, a, false, b, (short)0, acc[q], false, false);
        }
      }
#pragma unroll
      for (int q = 0; q < 4; ++q)
#pragma unroll
        for (int v = 0; v < 8; ++v) {
          const float t = acc[q][v] + cen[q][v] + bfv[q];
          hsum[q][v] += fmaxf(t, 0.0f);
        }
      if (more) scatterRow16(sA + ((k + 1) & 1) * 2048, mrow, c0, (const float*)cv);
      __syncthreads();
    }

    // ---- write h (f16) to LDS, row-major stride 264 ----
    {
      const int mb = (lane >> 4) * 8;
#pragma unroll
      for (int q = 0; q < 4; ++q) {
        const int N = wq * 64 + q * 16 + (lane & 15);
#pragma unroll
        for (int v = 0; v < 8; ++v) sH[(mb + v) * 264 + N] = (_Float16)hsum[q][v];
      }
    }
    __syncthreads();

    // ---- mu = h @ Wu + bu : K=256 (8 kt), waves split N=128 into 4x32 ----
    v8f mac[2];
    mac[0] = zero; mac[1] = zero;
    {
      const int mr = lane & 15;
      const int hb = (lane >> 4) * 8;
#pragma unroll
      for (int kt = 0; kt < 8; ++kt) {
        v16h a;
        const _Float16* arow = sH + mr * 264 + kt * 32 + hb;
#pragma unroll
        for (int e = 0; e < 8; ++e) { a[e] = arow[e]; a[8 + e] = arow[16 + e]; }
#pragma unroll
        for (int t2 = 0; t2 < 2; ++t2) {
          const int N = (wq * 2 + t2) * 16 + (lane & 15);
          v16h bW;
#pragma unroll
          for (int e = 0; e < 16; ++e) {
            const int Kv = kt * 32 + ((e >> 3) << 4) + hb + (e & 7);
            bW[e] = (_Float16)Wu[Kv * D_LAT + N];
          }
          mac[t2] = __builtin_amdgcn_wmma_f32_16x16x32_f16(
              false, a, false, bW, (short)0, mac[t2], false, false);
        }
      }
    }
    // store mu
    {
      const int mb = (lane >> 4) * 8;
#pragma unroll
      for (int t2 = 0; t2 < 2; ++t2) {
        const int N = (wq * 2 + t2) * 16 + (lane & 15);
        const float buv = bu[N];
#pragma unroll
        for (int v = 0; v < 8; ++v)
          mu_out[(size_t)(b0 + mb + v) * D_LAT + N] = mac[t2][v] + buv;
      }
    }
    // sigma = exp(h @ Wv + bv): 8 threads per row + shfl_xor tree reduction
    {
      const int row = tid >> 3;      // 0..15
      const int seg = tid & 7;       // 32-wide segment of the 256-dim dot
      float s = 0.f;
      const _Float16* hrow = sH + row * 264 + seg * 32;
      const float* wv = Wv + seg * 32;
#pragma unroll
      for (int j = 0; j < 32; ++j) s += (float)hrow[j] * wv[j];
      s += __shfl_xor(s, 1);
      s += __shfl_xor(s, 2);
      s += __shfl_xor(s, 4);
      if (seg == 0) sig_out[b0 + row] = expf(s + bv[0]);
    }
    __syncthreads();  // before next row tile reuses sCen/sA/sH
  }
}

extern "C" void kernel_launch(void* const* d_in, const int* in_sizes, int n_in,
                              void* d_out, int out_size, void* d_ws, size_t ws_size,
                              hipStream_t stream) {
  const int*   center = (const int*)  d_in[0];
  const int*   ctx    = (const int*)  d_in[1];
  const float* emb    = (const float*)d_in[2];
  const float* Wf     = (const float*)d_in[3];
  const float* bf     = (const float*)d_in[4];
  const float* Wu     = (const float*)d_in[5];
  const float* bu     = (const float*)d_in[6];
  const float* Wv     = (const float*)d_in[7];
  const float* bv     = (const float*)d_in[8];
  float* mu  = (float*)d_out;                    // (B, 128)
  float* sig = mu + (size_t)BATCH * D_LAT;       // (B,)

  dim3 grid(BATCH / (16 * ROWTILES));   // 512 blocks
  dim3 block(BLOCK);                    // 4 waves
  hipLaunchKernelGGL(enc_kernel, grid, block, 0, stream,
                     center, ctx, emb, Wf, bf, Wu, bu, Wv, bv, mu, sig);
}
